// Model_62173946577086
// MI455X (gfx1250) — matrix operand order repaired and checked by reading, not compiled, not run on hardware
//
#include <hip/hip_runtime.h>
#include <hip/hip_bf16.h>
#include <math.h>

#define T_STEPS 128
#define BATCH   256
#define DIN     256
#define DH      1024
#define DOUT    256

typedef _Float16 half_t;
typedef __attribute__((ext_vector_type(16))) _Float16 v16h;
typedef __attribute__((ext_vector_type(8)))  _Float16 v8h;
typedef __attribute__((ext_vector_type(8)))  float    v8f;

// ---------------------------------------------------------------------------
// Conversion kernels (one-time, feed the WMMA-friendly f16 layouts)
// ---------------------------------------------------------------------------

__global__ void cvt_f16_kernel(const float* __restrict__ in,
                               half_t* __restrict__ out, int n) {
    int i = blockIdx.x * blockDim.x + threadIdx.x;
    if (i < n) out[i] = (half_t)in[i];
}

// in: [K][N] f32 row-major  ->  out: [N][K] f16 row-major (B operand layout)
__global__ void transpose_f16_kernel(const float* __restrict__ in,
                                     half_t* __restrict__ out, int K, int N) {
    int idx = blockIdx.x * blockDim.x + threadIdx.x;
    if (idx < K * N) {
        int n = idx / K;
        int k = idx - n * K;
        out[idx] = (half_t)in[k * N + n];   // coalesced writes
    }
}

__global__ void zero_f16_kernel(half_t* __restrict__ p, int n) {
    int i = blockIdx.x * blockDim.x + threadIdx.x;
    if (i < n) p[i] = (half_t)0.0f;
}

// ---------------------------------------------------------------------------
// WMMA fragment loaders (contiguous k order: lane half h holds k0+16h..k0+16h+15, wave32)
// ---------------------------------------------------------------------------

// A: 16x32 tile of a row-major [M][K] f16 matrix.
// lane l (l<16): row m=l, K = {k0..k0+15}
// lane l (>=16): row m=l-16, K = {k0+16..k0+31}
__device__ __forceinline__ v16h load_a_frag(const half_t* __restrict__ A,
                                            int lda, int row0, int k0, int lane) {
    const half_t* p = A + (size_t)(row0 + (lane & 15)) * lda + k0 + ((lane >> 4) << 4);
    v8h lo = *(const v8h*)p;          // 16B contiguous
    v8h hi = *(const v8h*)(p + 8);    // 16B contiguous
    v16h a;
#pragma unroll
    for (int i = 0; i < 8; ++i) { a[i] = lo[i]; a[i + 8] = hi[i]; }
    return a;
}

// B: 32x16 tile, sourced from pre-transposed weights Bt = W^T stored [N][K].
// lane l: column n = l&15, K = k0 + (l>>4)*16 + i  (16 contiguous halves)
__device__ __forceinline__ v16h load_b_frag(const half_t* __restrict__ Bt,
                                            int ldk, int col0, int k0, int lane) {
    const half_t* p = Bt + (size_t)(col0 + (lane & 15)) * ldk + k0 + ((lane >> 4) << 4);
    return *(const v16h*)p;           // 32B contiguous
}

#define WMMA_F16(A, B, C) \
    __builtin_amdgcn_wmma_f32_16x16x32_f16(false, (A), false, (B), (short)0, (C), false, false)

// ---------------------------------------------------------------------------
// One recurrence step: hOut = tanh(x @ W1x + hIn @ W1h + b1)
// Each wave computes a 2x2 block of 16x16 tiles (32 rows x 32 cols):
//   per K-chunk: 4 A-halves + 4 B-halves (8 x b128) feed 4 WMMAs.
// grid: 32 blocks x 256 thr (8 waves) -> 256 wave-jobs = 1024 tiles
// ---------------------------------------------------------------------------
__global__ void __launch_bounds__(256)
rnn_step_kernel(const half_t* __restrict__ x16,    // [B][DIN]
                const half_t* __restrict__ hIn,    // [B][DH]
                const half_t* __restrict__ w1xT,   // [DH][DIN]
                const half_t* __restrict__ w1hT,   // [DH][DH]
                const float*  __restrict__ b1,
                half_t* __restrict__ hOut) {       // [B][DH]
    const int lane = threadIdx.x & 31;
    const int wave = threadIdx.x >> 5;
    const int job  = blockIdx.x * 8 + wave;   // 0..255
    const int mT   = (job & 7) << 5;          // 8 super-tiles over B=256
    const int nT   = (job >> 3) << 5;         // 32 super-tiles over DH=1024

    v8f acc00 = {}, acc01 = {}, acc10 = {}, acc11 = {};

    // x-projection: K = DIN = 256
#pragma unroll
    for (int k0 = 0; k0 < DIN; k0 += 32) {
        v16h a0 = load_a_frag(x16, DIN, mT,      k0, lane);
        v16h a1 = load_a_frag(x16, DIN, mT + 16, k0, lane);
        v16h b0 = load_b_frag(w1xT, DIN, nT,      k0, lane);
        v16h b1v = load_b_frag(w1xT, DIN, nT + 16, k0, lane);
        acc00 = WMMA_F16(a0, b0,  acc00);
        acc01 = WMMA_F16(a0, b1v, acc01);
        acc10 = WMMA_F16(a1, b0,  acc10);
        acc11 = WMMA_F16(a1, b1v, acc11);
    }
    // recurrence: K = DH = 1024
#pragma unroll 4
    for (int k0 = 0; k0 < DH; k0 += 32) {
        v16h a0 = load_a_frag(hIn, DH, mT,      k0, lane);
        v16h a1 = load_a_frag(hIn, DH, mT + 16, k0, lane);
        v16h b0 = load_b_frag(w1hT, DH, nT,      k0, lane);
        v16h b1v = load_b_frag(w1hT, DH, nT + 16, k0, lane);
        acc00 = WMMA_F16(a0, b0,  acc00);
        acc01 = WMMA_F16(a0, b1v, acc01);
        acc10 = WMMA_F16(a1, b0,  acc10);
        acc11 = WMMA_F16(a1, b1v, acc11);
    }

    // Epilogue: bias + tanh, store f16 row-major [B][DH]
    const int nlo = lane & 15;
    const int mrs = (lane >> 4) << 3;         // 0 or 8
    const float bv0 = b1[nT + nlo];
    const float bv1 = b1[nT + 16 + nlo];
#pragma unroll
    for (int r = 0; r < 8; ++r) {
        const int m0 = mT + mrs + r;
        const int m1 = mT + 16 + mrs + r;
        hOut[(size_t)m0 * DH + nT + nlo]      = (half_t)tanhf(acc00[r] + bv0);
        hOut[(size_t)m0 * DH + nT + 16 + nlo] = (half_t)tanhf(acc01[r] + bv1);
        hOut[(size_t)m1 * DH + nT + nlo]      = (half_t)tanhf(acc10[r] + bv0);
        hOut[(size_t)m1 * DH + nT + 16 + nlo] = (half_t)tanhf(acc11[r] + bv1);
    }
}

// ---------------------------------------------------------------------------
// Readout: out = h @ W2 + b2   (out f32 [B][DOUT])
// 2x2 tile blocking: 64 wave-jobs -> 8 blocks x 8 waves
// ---------------------------------------------------------------------------
__global__ void __launch_bounds__(256)
readout_kernel(const half_t* __restrict__ h,      // [B][DH]
               const half_t* __restrict__ w2T,    // [DOUT][DH]
               const float*  __restrict__ b2,
               float* __restrict__ out) {         // [B][DOUT]
    const int lane = threadIdx.x & 31;
    const int wave = threadIdx.x >> 5;
    const int job  = blockIdx.x * 8 + wave;   // 0..63
    const int mT   = (job & 7) << 5;          // 8 super-tiles over B
    const int nT   = (job >> 3) << 5;         // 8 super-tiles over DOUT

    v8f acc00 = {}, acc01 = {}, acc10 = {}, acc11 = {};
#pragma unroll 4
    for (int k0 = 0; k0 < DH; k0 += 32) {
        v16h a0 = load_a_frag(h, DH, mT,      k0, lane);
        v16h a1 = load_a_frag(h, DH, mT + 16, k0, lane);
        v16h b0 = load_b_frag(w2T, DH, nT,      k0, lane);
        v16h b1v = load_b_frag(w2T, DH, nT + 16, k0, lane);
        acc00 = WMMA_F16(a0, b0,  acc00);
        acc01 = WMMA_F16(a0, b1v, acc01);
        acc10 = WMMA_F16(a1, b0,  acc10);
        acc11 = WMMA_F16(a1, b1v, acc11);
    }

    const int nlo = lane & 15;
    const int mrs = (lane >> 4) << 3;
    const float c = b2[0];
#pragma unroll
    for (int r = 0; r < 8; ++r) {
        const int m0 = mT + mrs + r;
        const int m1 = mT + 16 + mrs + r;
        out[(size_t)m0 * DOUT + nT + nlo]      = acc00[r] + c;
        out[(size_t)m0 * DOUT + nT + 16 + nlo] = acc01[r] + c;
        out[(size_t)m1 * DOUT + nT + nlo]      = acc10[r] + c;
        out[(size_t)m1 * DOUT + nT + 16 + nlo] = acc11[r] + c;
    }
}

// ---------------------------------------------------------------------------
// Launcher
// ---------------------------------------------------------------------------
extern "C" void kernel_launch(void* const* d_in, const int* in_sizes, int n_in,
                              void* d_out, int out_size, void* d_ws, size_t ws_size,
                              hipStream_t stream) {
    const float* xs  = (const float*)d_in[0];   // [T][B][DIN]
    const float* W1x = (const float*)d_in[1];   // [DIN][DH]
    const float* W1h = (const float*)d_in[2];   // [DH][DH]
    const float* b1  = (const float*)d_in[3];   // [DH]
    const float* W2  = (const float*)d_in[4];   // [DH][DOUT]
    const float* b2  = (const float*)d_in[5];   // scalar
    float* out = (float*)d_out;                 // [B][DOUT]

    // Carve workspace (~20 MB)
    char* ws = (char*)d_ws;
    half_t* x16  = (half_t*)ws; ws += (size_t)T_STEPS * BATCH * DIN * sizeof(half_t);
    half_t* w1xT = (half_t*)ws; ws += (size_t)DH * DIN * sizeof(half_t);
    half_t* w1hT = (half_t*)ws; ws += (size_t)DH * DH * sizeof(half_t);
    half_t* w2T  = (half_t*)ws; ws += (size_t)DOUT * DH * sizeof(half_t);
    half_t* hA   = (half_t*)ws; ws += (size_t)BATCH * DH * sizeof(half_t);
    half_t* hB   = (half_t*)ws;

    // One-time conversions into WMMA-friendly layouts
    {
        int n = T_STEPS * BATCH * DIN;
        cvt_f16_kernel<<<(n + 255) / 256, 256, 0, stream>>>(xs, x16, n);
    }
    transpose_f16_kernel<<<(DIN * DH + 255) / 256, 256, 0, stream>>>(W1x, w1xT, DIN, DH);
    transpose_f16_kernel<<<(DH * DH + 255) / 256, 256, 0, stream>>>(W1h, w1hT, DH, DH);
    transpose_f16_kernel<<<(DH * DOUT + 255) / 256, 256, 0, stream>>>(W2, w2T, DH, DOUT);
    {
        int n = BATCH * DH;
        zero_f16_kernel<<<(n + 255) / 256, 256, 0, stream>>>(hA, n);
    }

    // Serial recurrence: one kernel per timestep (device-wide sync at launch
    // boundaries; all launches captured into the graph on `stream`).
    half_t* hc = hA;
    half_t* hn = hB;
    for (int t = 0; t < T_STEPS; ++t) {
        rnn_step_kernel<<<32, 256, 0, stream>>>(
            x16 + (size_t)t * BATCH * DIN, hc, w1xT, w1hT, b1, hn);
        half_t* tmp = hc; hc = hn; hn = tmp;
    }

    // Readout
    readout_kernel<<<8, 256, 0, stream>>>(hc, w2T, b2, out);
}
